// GraphDenoiseConvV5_37847251812411
// MI455X (gfx1250) — compile-verified, matrix operand order checked by reading, not run on hardware
//
#include <hip/hip_runtime.h>
#include <cstdint>

// ---------------------------------------------------------------------------
// Types for CDNA5 WMMA (wave32, v_wmma_f32_16x16x32_bf16)
// ---------------------------------------------------------------------------
typedef __attribute__((ext_vector_type(16))) __bf16 v16bf;
typedef __attribute__((ext_vector_type(8)))  float  v8f;

#define BB    8
#define KN    64
#define KK2   4096     // K*K
#define ECNT  2016     // K*(K-1)/2
#define LDK   40       // LDS row stride in bf16 elems (32 + 8 pad; 80B rows -> 16B aligned)

#if __has_builtin(__builtin_amdgcn_global_load_async_to_lds_b128)
#define HAVE_ASYNC_LDS 1
// 128-bit payload type matching the builtin's prototype (vector_size int)
typedef int b128i __attribute__((__vector_size__(16)));
typedef __attribute__((address_space(1))) b128i* gptr_b128;
typedef __attribute__((address_space(3))) b128i* lptr_b128;
#else
#define HAVE_ASYNC_LDS 0
#endif

__device__ __forceinline__ unsigned short f2bf(float x) {
  unsigned u = __float_as_uint(x);
  unsigned r = u + 0x7FFFu + ((u >> 16) & 1u);   // round-to-nearest-even
  return (unsigned short)(r >> 16);
}
__device__ __forceinline__ float bf2f(unsigned short h) {
  return __uint_as_float(((unsigned)h) << 16);
}
__device__ __forceinline__ unsigned pack2(float a, float b) {
  return (unsigned)f2bf(a) | ((unsigned)f2bf(b) << 16);
}

__device__ __forceinline__ void async_waitcnt0() {
#if __has_builtin(__builtin_amdgcn_s_wait_asynccnt)
  __builtin_amdgcn_s_wait_asynccnt(0);
#else
  asm volatile("s_wait_asynccnt 0x0" ::: "memory");
#endif
}

union Frag { v16bf v; unsigned u[8]; };

// Gather one 16-row x 32-K bf16 fragment (A layout; B identical with N for M).
// VGPR j holds K = ((j&4)<<2) + 8*g + 2*(j&3) + {0,1}, g = lane>>4 (ISA 7.12.2)
// Per fragment this is two contiguous 16B runs -> compiler emits 2x ds_load_b128.
__device__ __forceinline__ void frag_gather(const unsigned short* row, int g, Frag& f) {
#pragma unroll
  for (int j = 0; j < 8; ++j) {
    int kc = ((j & 4) << 2) + 8 * g + 2 * (j & 3);
    f.u[j] = *(const unsigned*)(row + kc);
  }
}

struct ATile { uint4 lo, hi; };     // 16 bf16, packed, ready for LDS b128 stores
struct WTile { float v[16]; };      // 4 k x 4 n sub-block; v[kk*4+nn]

// ---------------------------------------------------------------------------
// Generic batched GEMM:  C[b] (op)= act( pscale * (mapA(A[b]) @ W) + bias )
//   amode: 0 direct, 1 row>>6 (nf_self), 2 row&63 (nf_nb), 3 rowmap[] (tri)
//   ABF template: A is raw bf16 (1) or fp32 (0).  C fp32 or raw bf16 (cbf16).
//   bf16-A fast path uses GLOBAL_LOAD_ASYNC_TO_LDS_B128 (double-buffered sA).
// ---------------------------------------------------------------------------
struct GemmP {
  const void* A; long long sAb; int lda;
  const float* W; int ldw;
  const float* bias;
  void* C; long long sCb; int ldc; int cbf16;
  int M, N, Kd;
  int amode; const int* rowmap;
  float pscale;  // product scale (applied in epilogue)
  int acc;       // 1: C += result (bias should be null)
  int act;       // 1: leaky_relu(0.01) after accumulate
};

template <int ABF>
__global__ __launch_bounds__(128) void k_gemm(GemmP p) {
  __shared__ unsigned short sA[2][64 * LDK];
  __shared__ unsigned short sB[64 * LDK];
  const int zb  = blockIdx.z;
  const int m0  = blockIdx.y * 64;
  const int n0  = blockIdx.x * 64;
  const int tid = threadIdx.x;
  const int lane = tid & 31, wave = tid >> 5;
  const int wm = wave >> 1, wn = wave & 1;      // 2x2 waves -> 64x64 C tile (32x32/wave)

  const float*          Af = (const float*)p.A + (long long)zb * p.sAb;
  const unsigned short* Ah = (const unsigned short*)p.A + (long long)zb * p.sAb;

  // staging assignments: A: 64 rows x 32 k, 16 consecutive k per thread
  const int aR  = tid >> 1;
  const int aK0 = (tid & 1) * 16;
  // W: 32 k x 64 n, each thread owns a 4k x 4n sub-block
  const int wKq = (tid & 7) * 4;    // k base 0..28
  const int wNq = (tid >> 3) * 4;   // n base 0..60

  const int gr = m0 + aR;
  const bool rowOK = gr < p.M;
  int arow = 0;
  if (rowOK) {
    if      (p.amode == 0) arow = gr;
    else if (p.amode == 1) arow = gr >> 6;
    else if (p.amode == 2) arow = gr & 63;
    else                   arow = p.rowmap[gr];
  }
  const float*          aSrcF = Af + (long long)arow * p.lda;
  const unsigned short* aSrcH = Ah + (long long)arow * p.lda;

  // vectorization legality (uniform): 16B-aligned rows
  const bool avec = ABF ? ((((uintptr_t)p.A & 15) == 0) && ((p.lda & 7) == 0))
                        : ((((uintptr_t)p.A & 15) == 0) && ((p.lda & 3) == 0));
  const bool wvec = (((uintptr_t)p.W & 15) == 0) && ((p.ldw & 3) == 0);

  auto loadAt = [&](int k0) -> ATile {
    ATile t;
    const int kb = k0 + aK0;
    if (rowOK && avec && (kb + 16 <= p.Kd)) {
      if (ABF) {
        t.lo = *(const uint4*)(aSrcH + kb);
        t.hi = *(const uint4*)(aSrcH + kb + 8);
      } else {
        float4 x0 = *(const float4*)(aSrcF + kb);
        float4 x1 = *(const float4*)(aSrcF + kb + 4);
        float4 x2 = *(const float4*)(aSrcF + kb + 8);
        float4 x3 = *(const float4*)(aSrcF + kb + 12);
        t.lo.x = pack2(x0.x, x0.y); t.lo.y = pack2(x0.z, x0.w);
        t.lo.z = pack2(x1.x, x1.y); t.lo.w = pack2(x1.z, x1.w);
        t.hi.x = pack2(x2.x, x2.y); t.hi.y = pack2(x2.z, x2.w);
        t.hi.z = pack2(x3.x, x3.y); t.hi.w = pack2(x3.z, x3.w);
      }
    } else {
      // branch-free clamp + select fallback
      unsigned pk[8];
#pragma unroll
      for (int q = 0; q < 8; ++q) {
        int k0e = kb + 2 * q, k1e = kb + 2 * q + 1;
        int c0 = k0e < p.Kd ? k0e : (p.Kd - 1);
        int c1 = k1e < p.Kd ? k1e : (p.Kd - 1);
        unsigned short v0, v1;
        if (ABF) { v0 = aSrcH[c0]; v1 = aSrcH[c1]; }
        else     { v0 = f2bf(aSrcF[c0]); v1 = f2bf(aSrcF[c1]); }
        v0 = (rowOK && k0e < p.Kd) ? v0 : (unsigned short)0;
        v1 = (rowOK && k1e < p.Kd) ? v1 : (unsigned short)0;
        pk[q] = (unsigned)v0 | ((unsigned)v1 << 16);
      }
      t.lo.x = pk[0]; t.lo.y = pk[1]; t.lo.z = pk[2]; t.lo.w = pk[3];
      t.hi.x = pk[4]; t.hi.y = pk[5]; t.hi.z = pk[6]; t.hi.w = pk[7];
    }
    return t;
  };

  // Stage A tile directly into LDS buffer `buf` (async fast path for bf16 A).
  auto stageA = [&](int k0, unsigned short* buf) {
    const int kb = k0 + aK0;
    unsigned short* dst = &buf[aR * LDK + aK0];
#if HAVE_ASYNC_LDS
    if (ABF && rowOK && avec && (kb + 16 <= p.Kd)) {
      __builtin_amdgcn_global_load_async_to_lds_b128(
          (gptr_b128)(uintptr_t)(aSrcH + kb),
          (lptr_b128)(uintptr_t)dst, 0, 0);
      __builtin_amdgcn_global_load_async_to_lds_b128(
          (gptr_b128)(uintptr_t)(aSrcH + kb + 8),
          (lptr_b128)(uintptr_t)(dst + 8), 0, 0);
      return;
    }
#endif
    ATile t = loadAt(k0);
    *(uint4*)dst       = t.lo;
    *(uint4*)(dst + 8) = t.hi;
  };

  auto loadWt = [&](int k0) -> WTile {
    WTile t;
    const int kk0 = k0 + wKq;
    const int gn  = n0 + wNq;
    if (wvec && (kk0 + 4 <= p.Kd) && (gn + 4 <= p.N)) {
#pragma unroll
      for (int kk = 0; kk < 4; ++kk) {
        float4 x = *(const float4*)(p.W + (long long)(kk0 + kk) * p.ldw + gn);
        t.v[kk * 4 + 0] = x.x; t.v[kk * 4 + 1] = x.y;
        t.v[kk * 4 + 2] = x.z; t.v[kk * 4 + 3] = x.w;
      }
    } else {
#pragma unroll
      for (int kk = 0; kk < 4; ++kk) {
        int k = kk0 + kk; int kcl = k < p.Kd ? k : (p.Kd - 1);
#pragma unroll
        for (int nn = 0; nn < 4; ++nn) {
          int n = gn + nn; int ncl = n < p.N ? n : (p.N - 1);
          float f = p.W[(long long)kcl * p.ldw + ncl];
          t.v[kk * 4 + nn] = ((k < p.Kd) && (n < p.N)) ? f : 0.f;
        }
      }
    }
    return t;
  };

  auto storeW = [&](const WTile& t) {
#pragma unroll
    for (int nn = 0; nn < 4; ++nn) {
      uint2 u;
      u.x = pack2(t.v[0 * 4 + nn], t.v[1 * 4 + nn]);
      u.y = pack2(t.v[2 * 4 + nn], t.v[3 * 4 + nn]);
      *(uint2*)&sB[(wNq + nn) * LDK + wKq] = u;   // ds_store_b64
    }
  };

  constexpr bool ASY = (ABF != 0) && (HAVE_ASYNC_LDS != 0);

  v8f acc00 = {0.f,0.f,0.f,0.f,0.f,0.f,0.f,0.f};
  v8f acc01 = acc00, acc10 = acc00, acc11 = acc00;

  // software pipeline (register double-buffer, or async LDS double-buffer for bf16 A)
  ATile ra{};
  WTile rw;
  if (ASY) stageA(0, sA[0]);
  else     ra = loadAt(0);
  rw = loadWt(0);

  int bufi = 0;
  for (int k0 = 0; k0 < p.Kd; k0 += 32) {
    // ---- commit staged tile to LDS --------------------------------------
    if (!ASY) {
      unsigned short* dst = &sA[0][aR * LDK + aK0];
      *(uint4*)dst       = ra.lo;            // ds_store_b128
      *(uint4*)(dst + 8) = ra.hi;
    }
    storeW(rw);
    if (ASY) async_waitcnt0();               // this wave's async copies for tile k0 done

    // ---- issue next tile (in flight during WMMAs) ------------------------
    if (k0 + 32 < p.Kd) {
      if (ASY) stageA(k0 + 32, sA[bufi ^ 1]);
      else     ra = loadAt(k0 + 32);
      rw = loadWt(k0 + 32);
      if (k0 + 64 < p.Kd)   // warm L2 two tiles ahead
        __builtin_prefetch(&p.W[(long long)(k0 + 64 + wKq) * p.ldw + n0], 0, 0);
    }
    __syncthreads();

    // ---- compute ---------------------------------------------------------
    const unsigned short* sAc = sA[ASY ? bufi : 0];
    const int g = lane >> 4, r16 = lane & 15;
    Frag a0, a1, b0, b1;
    frag_gather(&sAc[(wm * 32 + r16) * LDK], g, a0);
    frag_gather(&sAc[(wm * 32 + 16 + r16) * LDK], g, a1);
    frag_gather(&sB[(wn * 32 + r16) * LDK], g, b0);
    frag_gather(&sB[(wn * 32 + 16 + r16) * LDK], g, b1);
    acc00 = __builtin_amdgcn_wmma_f32_16x16x32_bf16(false, a0.v, false, b0.v, (short)0, acc00, false, false);
    acc01 = __builtin_amdgcn_wmma_f32_16x16x32_bf16(false, a0.v, false, b1.v, (short)0, acc01, false, false);
    acc10 = __builtin_amdgcn_wmma_f32_16x16x32_bf16(false, a1.v, false, b0.v, (short)0, acc10, false, false);
    acc11 = __builtin_amdgcn_wmma_f32_16x16x32_bf16(false, a1.v, false, b1.v, (short)0, acc11, false, false);
    __syncthreads();
    bufi ^= 1;
  }

  // ---- epilogue: C/D layout: lanes0-15 N=lane, vgpr v -> M=v; lanes16-31 M=8+v
  const int g = lane >> 4, nl = lane & 15;
#pragma unroll
  for (int tm = 0; tm < 2; ++tm) {
#pragma unroll
    for (int tn = 0; tn < 2; ++tn) {
      v8f accv = tm ? (tn ? acc11 : acc10) : (tn ? acc01 : acc00);
      int col = n0 + wn * 32 + tn * 16 + nl;
      if (col >= p.N) continue;
      float bi = p.bias ? p.bias[col] : 0.f;
#pragma unroll
      for (int vv = 0; vv < 8; ++vv) {
        int row = m0 + wm * 32 + tm * 16 + g * 8 + vv;
        if (row >= p.M) continue;
        long long ci = (long long)zb * p.sCb + (long long)row * p.ldc + col;
        float val = accv[vv] * p.pscale + bi;
        if (p.acc) val += p.cbf16 ? bf2f(((const unsigned short*)p.C)[ci])
                                  : ((const float*)p.C)[ci];
        if (p.act) val = val > 0.f ? val : 0.01f * val;
        if (p.cbf16) ((unsigned short*)p.C)[ci] = f2bf(val);
        else         ((float*)p.C)[ci] = val;
      }
    }
  }
}

// ---------------------------------------------------------------------------
// Elementwise / graph kernels
// ---------------------------------------------------------------------------
__global__ __launch_bounds__(256) void k_tri(int* ti, int* tj, int* mf, int* ms) {
  int r = blockIdx.x * 256 + threadIdx.x;
  if (r >= ECNT) return;
  int rr = r, i = 0;
  while (rr >= 63 - i) { rr -= 63 - i; ++i; }   // row i has K-1-i entries
  int j = i + 1 + rr;
  ti[r] = i; tj[r] = j; mf[r] = i * KN + j; ms[r] = j * KN + i;
}

__global__ __launch_bounds__(256) void k_temb(const int* t, float* emb) {
  int idx = blockIdx.x * 256 + threadIdx.x;
  if (idx >= BB * 100) return;
  int b = idx / 100, c = idx % 100;
  float w = __powf(10000.f, -2.f * (float)(c & ~1) / 100.f);
  float x = (float)t[b] * w;
  emb[idx] = (c & 1) ? __cosf(x) : __sinf(x);
}

__global__ __launch_bounds__(256) void k_pemb(float* emb) {
  int idx = blockIdx.x * 256 + threadIdx.x;
  if (idx >= KN * 200) return;
  int n = idx / 200, c = idx % 200;
  float w = __powf(10000.f, -2.f * (float)(c & ~1) / 200.f);
  float x = (float)n * w;
  emb[idx] = (c & 1) ? __cosf(x) : __sinf(x);
}

__global__ __launch_bounds__(256) void k_zero(float* p, long long n) {
  long long i = (long long)blockIdx.x * 256 + threadIdx.x;
  if (i < n) p[i] = 0.f;
}

__global__ __launch_bounds__(256) void k_scatter_edges(const float* E, const int* ti,
                                                       const int* tj, float* Esym,
                                                       float* Edir) {
  int idx = blockIdx.x * 256 + threadIdx.x;
  if (idx >= BB * ECNT) return;
  int b = idx / ECNT, r = idx % ECNT;
  const float* e = E + ((long long)b * ECNT + r) * 13;
  int i = ti[r], j = tj[r];
  float* s1 = Esym + ((long long)b * KK2 + i * KN + j) * 10;
  float* s2 = Esym + ((long long)b * KK2 + j * KN + i) * 10;
#pragma unroll
  for (int c = 0; c < 10; ++c) { float v = e[3 + c]; s1[c] = v; s2[c] = v; }
  float* d1 = Edir + ((long long)b * KK2 + i * KN + j) * 3;
  float* d2 = Edir + ((long long)b * KK2 + j * KN + i) * 3;
  d1[0] = e[0]; d1[1] = e[1]; d1[2] = e[2];
  d2[0] = e[0]; d2[1] = e[2]; d2[2] = e[1];   // perm [0,2,1] on swapped side
}

__global__ __launch_bounds__(256) void k_valid(const float* Edir, float* valid) {
  int idx = blockIdx.x * 256 + threadIdx.x;
  if (idx >= BB * KK2) return;
  int r = idx % KK2; int i = r >> 6, j = r & 63;
  float d0 = Edir[(long long)idx * 3];
  valid[idx] = (i == j) ? 0.f : 1.f / (1.f + __expf(-(0.5f - d0) * 10.f));
}

// masked mean aggregation over neighbors j for one conv layer slice of ecat
__global__ __launch_bounds__(256) void k_agg(const unsigned short* ecat, int layer,
                                             const float* valid, float* agg) {
  int blk = blockIdx.x;                 // BB*KN blocks
  int b = blk / KN, i = blk % KN;
  const float* vrow = valid + (long long)b * KK2 + i * KN;
  float denom = 1e-6f;
#pragma unroll 8
  for (int j = 0; j < KN; ++j) denom += vrow[j];
  for (int c = threadIdx.x; c < 512; c += 256) {
    float s = 0.f;
    for (int j = 0; j < KN; ++j) {
      float ev = bf2f(ecat[((long long)b * KK2 + i * KN + j) * 3072 + layer * 512 + c]);
      s += ev * vrow[j];
    }
    agg[((long long)b * KN + i) * 512 + c] = s / denom;
  }
}

// src/dst node features -> [max,mean] (sym) and interleaved pair (dir)
__global__ __launch_bounds__(256) void k_pairfeat(const float* nf, const int* ti,
                                                  const int* tj, float* symn,
                                                  float* dirn) {
  int idx = blockIdx.x * 256 + threadIdx.x;
  if (idx >= BB * ECNT * 512) return;
  int c = idx % 512; int r = (idx / 512) % ECNT; int b = idx / (512 * ECNT);
  float s = nf[((long long)b * KN + ti[r]) * 512 + c];
  float d = nf[((long long)b * KN + tj[r]) * 512 + c];
  long long base = (long long)b * ECNT + r;
  symn[base * 1024 + c]         = fmaxf(s, d);
  symn[base * 1024 + 512 + c]   = 0.5f * (s + d);
  dirn[base * 1024 + 2 * c]     = s;
  dirn[base * 1024 + 2 * c + 1] = d;
}

// ---------------------------------------------------------------------------
// Host orchestration
// ---------------------------------------------------------------------------
static inline void gemm(hipStream_t s, int batches,
                        const void* A, long long sAb, int lda, int abf,
                        const float* W, int ldw, const float* bias,
                        void* C, long long sCb, int ldc, int cbf,
                        int M, int N, int Kd, int amode, const int* rowmap,
                        float pscale, int acc, int act) {
  GemmP p{A, sAb, lda, W, ldw, bias, C, sCb, ldc, cbf,
          M, N, Kd, amode, rowmap, pscale, acc, act};
  dim3 grid((N + 63) / 64, (M + 63) / 64, batches);
  if (abf) k_gemm<1><<<grid, dim3(128), 0, s>>>(p);
  else     k_gemm<0><<<grid, dim3(128), 0, s>>>(p);
}

extern "C" void kernel_launch(void* const* d_in, const int* in_sizes, int n_in,
                              void* d_out, int out_size, void* d_ws, size_t ws_size,
                              hipStream_t stream) {
  (void)in_sizes; (void)n_in; (void)out_size; (void)ws_size;
  const float* Vn = (const float*)d_in[0];   // (8,64,134)
  const float* En = (const float*)d_in[1];   // (8,2016,13)
  const int*   tP = (const int*)d_in[2];     // (8,)
  const float* P  = (const float*)d_in[3];   // flattened params (jax leaf order)

  // ---- parameter offsets (dicts sorted by key) -----------------------------
  size_t po = 0;
  auto take = [&](size_t n) { const float* q = P + po; po += n; return q; };
  // edge_dir_mlp : z=4099, hidden [512,256,128], wide=4995, out 3
  const float* dW1 = take((size_t)4099 * 512); const float* db1 = take(512);
  const float* dW2 = take((size_t)512 * 256);  const float* db2 = take(256);
  const float* dW3 = take((size_t)256 * 128);  const float* db3 = take(128);
  const float* dWo = take((size_t)4995 * 3);   const float* dbo = take(3);
  // edge_sym_mlp : z=4106, wide=5002, out 10
  const float* sW1 = take((size_t)4106 * 512); const float* sb1 = take(512);
  const float* sW2 = take((size_t)512 * 256);  const float* sb2 = take(256);
  const float* sW3 = take((size_t)256 * 128);  const float* sb3 = take(128);
  const float* sWo = take((size_t)5002 * 10);  const float* sbo = take(10);
  // layers (keys sorted: ec1,ec2,ed,es,pe1,pe2,te1,te2,v0,v1,v2)
  struct LW {
    const float *ec1W,*ec1b,*ec2W,*ec2b,*edW,*edb,*esW,*esb;
    const float *pe1W,*pe1b,*pe2W,*pe2b,*te1W,*te1b,*te2W,*te2b;
    const float *v0W,*v0b,*v1W,*v1b,*v2W,*v2b; int cin;
  } L[6];
  int cin = 134;
  for (int l = 0; l < 6; ++l) {
    L[l].cin  = cin;
    L[l].ec1W = take((size_t)1024 * 512); L[l].ec1b = take(512);
    L[l].ec2W = take((size_t)512 * 512);  L[l].ec2b = take(512);
    L[l].edW  = take((size_t)3 * 256);    L[l].edb  = take(256);
    L[l].esW  = take((size_t)10 * 256);   L[l].esb  = take(256);
    L[l].pe1W = take((size_t)200 * 200);  L[l].pe1b = take(200);
    L[l].pe2W = take((size_t)200 * 200);  L[l].pe2b = take(200);
    L[l].te1W = take((size_t)100 * 100);  L[l].te1b = take(100);
    L[l].te2W = take((size_t)100 * 100);  L[l].te2b = take(100);
    L[l].v0W  = take((size_t)(cin + 200) * 256); L[l].v0b = take(256);
    L[l].v1W  = take((size_t)356 * 256);  L[l].v1b  = take(256);
    L[l].v2W  = take((size_t)(512 + cin) * 512); L[l].v2b = take(512);
    cin = 512;
  }
  // node_mlp : z=512, hidden [512,256,256], wide=1536, out 134
  const float* nW1 = take((size_t)512 * 512);  const float* nb1 = take(512);
  const float* nW2 = take((size_t)512 * 256);  const float* nb2 = take(256);
  const float* nW3 = take((size_t)256 * 256);  const float* nb3 = take(256);
  const float* nWo = take((size_t)1536 * 134); const float* nbo = take(134);

  // ---- workspace bump allocator -------------------------------------------
  char* wsp = (char*)d_ws;
  size_t wo = 0;
  auto walloc = [&](size_t bytes) {
    void* q = wsp + wo;
    wo = (wo + bytes + 255) & ~(size_t)255;
    return q;
  };
  int*   ti   = (int*)walloc(ECNT * 4);
  int*   tj   = (int*)walloc(ECNT * 4);
  int*   mf   = (int*)walloc(ECNT * 4);     // i*K+j
  int*   ms   = (int*)walloc(ECNT * 4);     // j*K+i
  float* temb = (float*)walloc((size_t)BB * 100 * 4);
  float* tt1  = (float*)walloc((size_t)BB * 100 * 4);
  float* pt   = (float*)walloc((size_t)BB * 100 * 4);
  float* pemb = (float*)walloc((size_t)KN * 200 * 4);
  float* pp1  = (float*)walloc((size_t)KN * 200 * 4);
  float* pp   = (float*)walloc((size_t)KN * 200 * 4);
  float* Esym = (float*)walloc((size_t)BB * KK2 * 10 * 4);
  float* Edir = (float*)walloc((size_t)BB * KK2 * 3 * 4);
  float* valid= (float*)walloc((size_t)BB * KK2 * 4);
  float* nodeA= (float*)walloc((size_t)BB * KN * 512 * 4);
  float* nodeB= (float*)walloc((size_t)BB * KN * 512 * 4);
  float* nf1  = (float*)walloc((size_t)BB * KN * 256 * 4);
  float* nf2  = (float*)walloc((size_t)BB * KN * 256 * 4);
  float* esO  = (float*)walloc((size_t)BB * KK2 * 256 * 4);
  float* edO  = (float*)walloc((size_t)BB * KK2 * 256 * 4);
  float* efP  = (float*)walloc((size_t)BB * KK2 * 512 * 4);
  float* agg  = (float*)walloc((size_t)BB * KN * 512 * 4);
  unsigned short* ecat = (unsigned short*)walloc((size_t)BB * KK2 * 3072 * 2);
  float* symn = (float*)walloc((size_t)BB * ECNT * 1024 * 4);
  float* dirn = (float*)walloc((size_t)BB * ECNT * 1024 * 4);
  float* h1   = (float*)walloc((size_t)BB * ECNT * 512 * 4);
  float* h2   = (float*)walloc((size_t)BB * ECNT * 256 * 4);
  float* h3   = (float*)walloc((size_t)BB * ECNT * 128 * 4);
  float* nh1  = (float*)walloc((size_t)BB * KN * 512 * 4);
  float* nh2  = (float*)walloc((size_t)BB * KN * 256 * 4);
  float* nh3  = (float*)walloc((size_t)BB * KN * 256 * 4);

  // ---- static graph structure & embeddings --------------------------------
  k_tri<<<(ECNT + 255) / 256, 256, 0, stream>>>(ti, tj, mf, ms);
  k_temb<<<(BB * 100 + 255) / 256, 256, 0, stream>>>(tP, temb);
  k_pemb<<<(KN * 200 + 255) / 256, 256, 0, stream>>>(pemb);
  k_zero<<<(unsigned)(((long long)BB * KK2 * 10 + 255) / 256), 256, 0, stream>>>(Esym, (long long)BB * KK2 * 10);
  k_zero<<<(unsigned)(((long long)BB * KK2 * 3 + 255) / 256), 256, 0, stream>>>(Edir, (long long)BB * KK2 * 3);
  k_scatter_edges<<<(BB * ECNT + 255) / 256, 256, 0, stream>>>(En, ti, tj, Esym, Edir);
  k_valid<<<(BB * KK2 + 255) / 256, 256, 0, stream>>>(Edir, valid);

  // ---- conv layers ---------------------------------------------------------
  const void* nodeIn = (const void*)Vn;
  int       lda_n = 134;
  long long sA_n  = (long long)KN * 134;
  float* bufs[2] = {nodeA, nodeB};
  int cur = 0;
  long long sKK256 = (long long)KK2 * 256, sKK512 = (long long)KK2 * 512;
  long long s64x256 = (long long)KN * 256, s64x512 = (long long)KN * 512;

  for (int l = 0; l < 6; ++l) {
    const LW& W = L[l]; int ci = W.cin;
    // time / position embedding MLPs (per-layer weights)
    gemm(stream, 1, temb, 0, 100, 0, W.te1W, 100, W.te1b, tt1, 0, 100, 0, BB, 100, 100, 0, nullptr, 1.f, 0, 1);
    gemm(stream, 1, tt1,  0, 100, 0, W.te2W, 100, W.te2b, pt,  0, 100, 0, BB, 100, 100, 0, nullptr, 1.f, 0, 0);
    gemm(stream, 1, pemb, 0, 200, 0, W.pe1W, 200, W.pe1b, pp1, 0, 200, 0, KN, 200, 200, 0, nullptr, 1.f, 0, 1);
    gemm(stream, 1, pp1,  0, 200, 0, W.pe2W, 200, W.pe2b, pp,  0, 200, 0, KN, 200, 200, 0, nullptr, 1.f, 0, 0);
    // v0: lrelu([node_f, pe_p] @ W)
    gemm(stream, BB, nodeIn, sA_n, lda_n, 0, W.v0W, 256, W.v0b, nf1, s64x256, 256, 0, KN, 256, ci, 0, nullptr, 1.f, 0, 0);
    gemm(stream, BB, pp, 0, 200, 0, W.v0W + (size_t)ci * 256, 256, nullptr, nf1, s64x256, 256, 0, KN, 256, 200, 0, nullptr, 1.f, 1, 1);
    // v1: lrelu([nf, pe_t] @ W)   (pe_t broadcast over K via amode1)
    gemm(stream, BB, nf1, s64x256, 256, 0, W.v1W, 256, W.v1b, nf2, s64x256, 256, 0, KN, 256, 256, 0, nullptr, 1.f, 0, 0);
    gemm(stream, BB, pt, 100, 100, 0, W.v1W + (size_t)256 * 256, 256, nullptr, nf2, s64x256, 256, 0, KN, 256, 100, 1, nullptr, 1.f, 1, 1);
    // es / ed
    gemm(stream, BB, Esym, (long long)KK2 * 10, 10, 0, W.esW, 256, W.esb, esO, sKK256, 256, 0, KK2, 256, 10, 0, nullptr, 1.f, 0, 1);
    gemm(stream, BB, Edir, (long long)KK2 * 3,  3,  0, W.edW, 256, W.edb, edO, sKK256, 256, 0, KK2, 256, 3,  0, nullptr, 1.f, 0, 1);
    // ec1: [es, ed, nf_self, nf_nb] @ W  (broadcasts via amode1/2)
    gemm(stream, BB, esO, sKK256, 256, 0, W.ec1W,                       512, W.ec1b,  efP, sKK512, 512, 0, KK2, 512, 256, 0, nullptr, 1.f, 0, 0);
    gemm(stream, BB, edO, sKK256, 256, 0, W.ec1W + (size_t)256 * 512,   512, nullptr, efP, sKK512, 512, 0, KK2, 512, 256, 0, nullptr, 1.f, 1, 0);
    gemm(stream, BB, nf2, s64x256, 256, 0, W.ec1W + (size_t)512 * 512,  512, nullptr, efP, sKK512, 512, 0, KK2, 512, 256, 1, nullptr, 1.f, 1, 0);
    gemm(stream, BB, nf2, s64x256, 256, 0, W.ec1W + (size_t)768 * 512,  512, nullptr, efP, sKK512, 512, 0, KK2, 512, 256, 2, nullptr, 1.f, 1, 1);
    // ec2 -> bf16 edge_cat slice (layer l occupies cols [l*512, l*512+512))
    gemm(stream, BB, efP, sKK512, 512, 0, W.ec2W, 512, W.ec2b, ecat + (size_t)l * 512,
         (long long)KK2 * 3072, 3072, 1, KK2, 512, 512, 0, nullptr, 1.f, 0, 1);
    // masked mean aggregation
    k_agg<<<BB * KN, 256, 0, stream>>>(ecat, l, valid, agg);
    // v2: lrelu([node_f, agg] @ W)
    float* nodeNxt = bufs[cur];
    gemm(stream, BB, nodeIn, sA_n, lda_n, 0, W.v2W, 512, W.v2b, nodeNxt, s64x512, 512, 0, KN, 512, ci, 0, nullptr, 1.f, 0, 0);
    gemm(stream, BB, agg, s64x512, 512, 0, W.v2W + (size_t)ci * 512, 512, nullptr, nodeNxt, s64x512, 512, 0, KN, 512, 512, 0, nullptr, 1.f, 1, 1);
    nodeIn = nodeNxt; lda_n = 512; sA_n = s64x512; cur ^= 1;
  }
  const float* nodeF = (const float*)nodeIn;

  // ---- node head ----------------------------------------------------------
  gemm(stream, BB, nodeF, s64x512, 512, 0, nW1, 512, nb1, nh1, s64x512, 512, 0, KN, 512, 512, 0, nullptr, 1.f, 0, 1);
  gemm(stream, BB, nh1, s64x512, 512, 0, nW2, 256, nb2, nh2, s64x256, 256, 0, KN, 256, 512, 0, nullptr, 1.f, 0, 1);
  gemm(stream, BB, nh2, s64x256, 256, 0, nW3, 256, nb3, nh3, s64x256, 256, 0, KN, 256, 256, 0, nullptr, 1.f, 0, 1);
  float* vpred = (float*)d_out;                     // (8,64,134)
  long long sVp = (long long)KN * 134;
  gemm(stream, BB, nodeF, s64x512, 512, 0, nWo,                       134, nbo,     vpred, sVp, 134, 0, KN, 134, 512, 0, nullptr, 1.f, 0, 0);
  gemm(stream, BB, nh1, s64x512, 512, 0, nWo + (size_t)512 * 134,     134, nullptr, vpred, sVp, 134, 0, KN, 134, 512, 0, nullptr, 1.f, 1, 0);
  gemm(stream, BB, nh2, s64x256, 256, 0, nWo + (size_t)1024 * 134,    134, nullptr, vpred, sVp, 134, 0, KN, 134, 256, 0, nullptr, 1.f, 1, 0);
  gemm(stream, BB, nh3, s64x256, 256, 0, nWo + (size_t)1280 * 134,    134, nullptr, vpred, sVp, 134, 0, KN, 134, 256, 0, nullptr, 1.f, 1, 0);

  // ---- edge pair node features -------------------------------------------
  k_pairfeat<<<(unsigned)(((long long)BB * ECNT * 512 + 255) / 256), 256, 0, stream>>>(nodeF, ti, tj, symn, dirn);

  long long sE1024 = (long long)ECNT * 1024, sE512 = (long long)ECNT * 512;
  long long sE256 = (long long)ECNT * 256, sE128 = (long long)ECNT * 128;
  long long sCat = (long long)KK2 * 3072;
  long long sEsym = (long long)KK2 * 10, sEdir = (long long)KK2 * 3;

  // ---- edge_sym head: x = [symn(1024), (cat+catT)/2 (3072), Esym(10)] ------
  gemm(stream, BB, symn, sE1024, 1024, 0, sW1,                        512, sb1,     h1, sE512, 512, 0, ECNT, 512, 1024, 0, nullptr, 1.0f, 0, 0);
  gemm(stream, BB, ecat, sCat, 3072, 1, sW1 + (size_t)1024 * 512,     512, nullptr, h1, sE512, 512, 0, ECNT, 512, 3072, 3, mf, 0.5f, 1, 0);
  gemm(stream, BB, ecat, sCat, 3072, 1, sW1 + (size_t)1024 * 512,     512, nullptr, h1, sE512, 512, 0, ECNT, 512, 3072, 3, ms, 0.5f, 1, 0);
  gemm(stream, BB, Esym, sEsym, 10, 0, sW1 + (size_t)4096 * 512,      512, nullptr, h1, sE512, 512, 0, ECNT, 512, 10,   3, mf, 1.0f, 1, 1);
  gemm(stream, BB, h1, sE512, 512, 0, sW2, 256, sb2, h2, sE256, 256, 0, ECNT, 256, 512, 0, nullptr, 1.f, 0, 1);
  gemm(stream, BB, h2, sE256, 256, 0, sW3, 128, sb3, h3, sE128, 128, 0, ECNT, 128, 256, 0, nullptr, 1.f, 0, 1);
  float* spred = (float*)d_out + (size_t)BB * KN * 134;   // (8,2016,10)
  long long sSp = (long long)ECNT * 10;
  gemm(stream, BB, symn, sE1024, 1024, 0, sWo,                        10, sbo,     spred, sSp, 10, 0, ECNT, 10, 1024, 0, nullptr, 1.0f, 0, 0);
  gemm(stream, BB, ecat, sCat, 3072, 1, sWo + (size_t)1024 * 10,      10, nullptr, spred, sSp, 10, 0, ECNT, 10, 3072, 3, mf, 0.5f, 1, 0);
  gemm(stream, BB, ecat, sCat, 3072, 1, sWo + (size_t)1024 * 10,      10, nullptr, spred, sSp, 10, 0, ECNT, 10, 3072, 3, ms, 0.5f, 1, 0);
  gemm(stream, BB, Esym, sEsym, 10, 0, sWo + (size_t)4096 * 10,       10, nullptr, spred, sSp, 10, 0, ECNT, 10, 10,   3, mf, 1.0f, 1, 0);
  gemm(stream, BB, h1, sE512, 512, 0, sWo + (size_t)4106 * 10,        10, nullptr, spred, sSp, 10, 0, ECNT, 10, 512,  0, nullptr, 1.f, 1, 0);
  gemm(stream, BB, h2, sE256, 256, 0, sWo + (size_t)4618 * 10,        10, nullptr, spred, sSp, 10, 0, ECNT, 10, 256,  0, nullptr, 1.f, 1, 0);
  gemm(stream, BB, h3, sE128, 128, 0, sWo + (size_t)4874 * 10,        10, nullptr, spred, sSp, 10, 0, ECNT, 10, 128,  0, nullptr, 1.f, 1, 0);

  // ---- edge_dir head: x = [dirn(1024), Edir(3), cat(3072)] ----------------
  gemm(stream, BB, dirn, sE1024, 1024, 0, dW1,                        512, db1,     h1, sE512, 512, 0, ECNT, 512, 1024, 0, nullptr, 1.0f, 0, 0);
  gemm(stream, BB, Edir, sEdir, 3, 0, dW1 + (size_t)1024 * 512,       512, nullptr, h1, sE512, 512, 0, ECNT, 512, 3,    3, mf, 1.0f, 1, 0);
  gemm(stream, BB, ecat, sCat, 3072, 1, dW1 + (size_t)1027 * 512,     512, nullptr, h1, sE512, 512, 0, ECNT, 512, 3072, 3, mf, 1.0f, 1, 1);
  gemm(stream, BB, h1, sE512, 512, 0, dW2, 256, db2, h2, sE256, 256, 0, ECNT, 256, 512, 0, nullptr, 1.f, 0, 1);
  gemm(stream, BB, h2, sE256, 256, 0, dW3, 128, db3, h3, sE128, 128, 0, ECNT, 128, 256, 0, nullptr, 1.f, 0, 1);
  float* dpred = (float*)d_out + (size_t)BB * KN * 134 + (size_t)BB * ECNT * 10;  // (8,2016,3)
  long long sDp = (long long)ECNT * 3;
  gemm(stream, BB, dirn, sE1024, 1024, 0, dWo,                        3, dbo,     dpred, sDp, 3, 0, ECNT, 3, 1024, 0, nullptr, 1.0f, 0, 0);
  gemm(stream, BB, Edir, sEdir, 3, 0, dWo + (size_t)1024 * 3,         3, nullptr, dpred, sDp, 3, 0, ECNT, 3, 3,    3, mf, 1.0f, 1, 0);
  gemm(stream, BB, ecat, sCat, 3072, 1, dWo + (size_t)1027 * 3,       3, nullptr, dpred, sDp, 3, 0, ECNT, 3, 3072, 3, mf, 1.0f, 1, 0);
  gemm(stream, BB, h1, sE512, 512, 0, dWo + (size_t)4099 * 3,         3, nullptr, dpred, sDp, 3, 0, ECNT, 3, 512,  0, nullptr, 1.f, 1, 0);
  gemm(stream, BB, h2, sE256, 256, 0, dWo + (size_t)4611 * 3,         3, nullptr, dpred, sDp, 3, 0, ECNT, 3, 256,  0, nullptr, 1.f, 1, 0);
  gemm(stream, BB, h3, sE128, 128, 0, dWo + (size_t)4867 * 3,         3, nullptr, dpred, sDp, 3, 0, ECNT, 3, 128,  0, nullptr, 1.f, 1, 0);
}
